// HeadNetwork_45784351375628
// MI455X (gfx1250) — matrix operand order repaired and checked by reading, not compiled, not run on hardware
//
#include <hip/hip_runtime.h>

// ---------------------------------------------------------------------------
// CDNA5 / gfx1250 fused CenterPoint-style head network.
//   preds: 6x [conv3x3(256->256) -> BN -> ReLU -> conv1x1(256->nout)] -> 14ch
//   tgts : 4ch gaussian heatmap + 11ch scattered scalars
//   out  : (4, 29, 200, 200) f32
// Conv path: implicit GEMM on V_WMMA_F32_16X16X32_BF16, K = 9*256.
// Each wave: 2 M-tiles x 2 output rows (4 accumulators) so every A/B
// fragment load feeds two WMMAs.
// ---------------------------------------------------------------------------

typedef __attribute__((ext_vector_type(8)))  float  v8f;
typedef __attribute__((ext_vector_type(16))) __bf16 v16bf;

#define Bn   4
#define Cn   256
#define Hn   200
#define Wn   200
#define NBOX 128

// ---------------------------------------------------------------------------
// P1: NCHW f32 -> NHWC bf16 (tiled transpose through LDS for coalescing)
// grid: (ceil(W/32)=7, C/32=8, B*H=800), block (32,8)
// ---------------------------------------------------------------------------
__global__ void k_nchw_to_nhwc_bf16(const float* __restrict__ in,
                                    __bf16* __restrict__ outp) {
  __shared__ float t[32][33];
  const int xt = blockIdx.x, ct = blockIdx.y;
  const int bz = blockIdx.z;
  const int b = bz / Hn, y = bz % Hn;
  const int tx = threadIdx.x, ty = threadIdx.y;
  #pragma unroll
  for (int k = 0; k < 4; ++k) {
    int ch = ct * 32 + ty + k * 8;
    int x  = xt * 32 + tx;
    float v = 0.0f;
    if (x < Wn) v = in[(((size_t)b * Cn + ch) * Hn + y) * Wn + x];
    t[ty + k * 8][tx] = v;
  }
  __syncthreads();
  #pragma unroll
  for (int k = 0; k < 4; ++k) {
    int x  = xt * 32 + ty + k * 8;
    int ch = ct * 32 + tx;
    if (x < Wn)
      outp[((((size_t)b * Hn + y) * Wn + x) << 8) + ch] = (__bf16)t[tx][ty + k * 8];
  }
}

// ---------------------------------------------------------------------------
// P2: repack w1 (6,256,256,3,3) f32 into bf16 WMMA A-fragments:
//   wfrag[head][mtile(16)][tap(9)][kc(8)][lane(32)][e(16)]
// Per-lane element order (contiguous K, matching the B-fragment loads):
//   lane<16 : M=lane,    K=e
//   lane>=16: M=lane-16, K=16+e
// ---------------------------------------------------------------------------
__global__ void k_pack_w1(const float* __restrict__ w1,
                          __bf16* __restrict__ wfrag, int total) {
  int idx = blockIdx.x * blockDim.x + threadIdx.x;
  if (idx >= total) return;
  int t = idx;
  int e    = t & 15;  t >>= 4;
  int lane = t & 31;  t >>= 5;
  int kc   = t & 7;   t >>= 3;
  int tap  = t % 9;   t /= 9;
  int mt   = t & 15;  t >>= 4;
  int h    = t;
  int half = lane >> 4;
  int mrow = lane & 15;
  int K    = half * 16 + e;
  int inch = kc * 32 + K;
  int o    = mt * 16 + mrow;
  float v  = w1[(((size_t)h * Cn + o) * Cn + inch) * 9 + tap];
  wfrag[idx] = (__bf16)v;
}

// ---------------------------------------------------------------------------
// P3: fold BN into per-channel scale/shift
// ---------------------------------------------------------------------------
__global__ void k_bn_fold(const float* __restrict__ g, const float* __restrict__ be,
                          const float* __restrict__ m, const float* __restrict__ v,
                          float* __restrict__ scale, float* __restrict__ shift,
                          int total) {
  int i = blockIdx.x * blockDim.x + threadIdx.x;
  if (i >= total) return;
  float s = g[i] * rsqrtf(v[i] + 1e-5f);
  scale[i] = s;
  shift[i] = be[i] - m[i] * s;
}

// ---------------------------------------------------------------------------
// Main fused conv kernel. One workgroup = (head, b, row pair y0..y0+1,
// 16-pixel x-tile). 8 waves; wave w owns M-tiles {2w, 2w+1} for BOTH rows.
// K-loop = 9 taps x 8 chan-chunks; 4 WMMAs per step per wave.
// grid: (13, 100, 24), block 256
// ---------------------------------------------------------------------------
__global__ __launch_bounds__(256) void k_conv_heads(
    const __bf16* __restrict__ act,    // NHWC bf16
    const __bf16* __restrict__ wfrag,  // packed A fragments
    const float*  __restrict__ bnscale, const float* __restrict__ bnshift,
    const float*  __restrict__ w2, const float* __restrict__ b2,
    float* __restrict__ out) {
  static const int kNout[6] = {4, 2, 1, 3, 2, 2};
  static const int kOff[6]  = {0, 4, 6, 7, 10, 12};

  const int xt   = blockIdx.x;
  const int y0   = blockIdx.y * 2;
  const int b    = blockIdx.z / 6;
  const int head = blockIdx.z % 6;
  const int x0   = xt * 16;
  const int nout = kNout[head];
  const int obase = kOff[head];

  // patch rows y0-1 .. y0+2 (4 rows) x 18 cols x 256 ch, bf16 = 36864 B
  __shared__ __attribute__((aligned(32))) __bf16 patch[4][18][Cn];
  __shared__ float hbuf[2][16][Cn + 1];   // 32896 B
  __shared__ float sscale[Cn], sshift[Cn];
  __shared__ float w2s[4 * Cn];
  __shared__ float b2s[4];

  const int tid = threadIdx.x;

  // stage folded BN params and 1x1 weights for this head
  sscale[tid] = bnscale[head * Cn + tid];
  sshift[tid] = bnshift[head * Cn + tid];
  for (int i = tid; i < nout * Cn; i += 256) w2s[i] = w2[obase * Cn + i];
  if (tid < nout) b2s[tid] = b2[obase + tid];

  // stage 4x18x256 bf16 activation patch (zero-padded halo), 16B chunks
  for (int ci = tid; ci < 72 * 32; ci += 256) {
    int slot = ci >> 5;        // 0..71 = r*18 + c
    int sub  = ci & 31;        // 16B chunk inside the 512B channel row
    int r = slot / 18, c = slot % 18;
    int yy = y0 + r - 1, xx = x0 + c - 1;
    uint4 v = make_uint4(0u, 0u, 0u, 0u);
    if (yy >= 0 && yy < Hn && xx >= 0 && xx < Wn)
      v = *(const uint4*)(act + ((((size_t)b * Hn + yy) * Wn + xx) << 8) + (sub << 3));
    *(uint4*)((char*)(&patch[r][c][0]) + sub * 16) = v;
  }
  __syncthreads();

  const int lane = tid & 31;
  const int wave = tid >> 5;
  const int half = lane >> 4;
  const int n    = lane & 15;       // pixel within tile / B-matrix column
  const int mt0  = wave * 2;
  const int mt1  = wave * 2 + 1;

  v8f acc00 = {}, acc01 = {};       // mt0 x {row0, row1}
  v8f acc10 = {}, acc11 = {};       // mt1 x {row0, row1}
  const __bf16* wbase = wfrag + (size_t)head * 16 * 9 * 8 * 32 * 16;
  const size_t lane_off = (size_t)lane * 16;

  for (int tap = 0; tap < 9; ++tap) {
    const int dr = tap / 3, dc = tap - dr * 3;
    const __bf16* a0base = wbase + (((size_t)mt0 * 9 + tap) * 8) * 32 * 16;
    const __bf16* a1base = wbase + (((size_t)mt1 * 9 + tap) * 8) * 32 * 16;
    if (tap < 8) {  // stream-prefetch next tap's weight fragments
      __builtin_prefetch(wbase + (((size_t)mt0 * 9 + tap + 1) * 8) * 32 * 16 + lane_off, 0, 1);
      __builtin_prefetch(wbase + (((size_t)mt1 * 9 + tap + 1) * 8) * 32 * 16 + lane_off, 0, 1);
    }
    #pragma unroll
    for (int kc = 0; kc < 8; ++kc) {
      // B fragments: rows y0+dr-1 (for out row y0) and y0+dr (for y0+1)
      const v16bf bf0 = *(const v16bf*)&patch[dr]    [n + dc][kc * 32 + half * 16];
      const v16bf bf1 = *(const v16bf*)&patch[dr + 1][n + dc][kc * 32 + half * 16];
      const v16bf a0  = *(const v16bf*)(a0base + (size_t)kc * 32 * 16 + lane_off);
      const v16bf a1  = *(const v16bf*)(a1base + (size_t)kc * 32 * 16 + lane_off);
      acc00 = __builtin_amdgcn_wmma_f32_16x16x32_bf16(false, a0, false, bf0,
                                                      (short)0, acc00, false, false);
      acc01 = __builtin_amdgcn_wmma_f32_16x16x32_bf16(false, a0, false, bf1,
                                                      (short)0, acc01, false, false);
      acc10 = __builtin_amdgcn_wmma_f32_16x16x32_bf16(false, a1, false, bf0,
                                                      (short)0, acc10, false, false);
      acc11 = __builtin_amdgcn_wmma_f32_16x16x32_bf16(false, a1, false, bf1,
                                                      (short)0, acc11, false, false);
    }
  }

  // epilogue: BN + ReLU, spill h(256ch x 2rows x 16px) to LDS
  // C/D layout: VGPR i -> M = i + 8*half (+ mtile*16), N = lane&15
  #pragma unroll
  for (int i = 0; i < 8; ++i) {
    int c0 = mt0 * 16 + i + half * 8;
    int c1 = mt1 * 16 + i + half * 8;
    hbuf[0][n][c0] = fmaxf(acc00[i] * sscale[c0] + sshift[c0], 0.0f);
    hbuf[1][n][c0] = fmaxf(acc01[i] * sscale[c0] + sshift[c0], 0.0f);
    hbuf[0][n][c1] = fmaxf(acc10[i] * sscale[c1] + sshift[c1], 0.0f);
    hbuf[1][n][c1] = fmaxf(acc11[i] * sscale[c1] + sshift[c1], 0.0f);
  }
  __syncthreads();

  // 1x1 conv: thread (o, row, px) reduces 256 channels from LDS
  if (tid < nout * 32) {
    int o   = tid >> 5;
    int rem = tid & 31;
    int r   = rem >> 4;
    int px  = rem & 15;
    int x   = x0 + px;
    if (x < Wn) {
      float acc = b2s[o];
      const float* wrow = &w2s[o * Cn];
      const float* hrow = &hbuf[r][px][0];
      #pragma unroll 8
      for (int ch = 0; ch < Cn; ++ch) acc += hrow[ch] * wrow[ch];
      out[(((size_t)b * 29 + (obase + o)) * Hn + (y0 + r)) * Wn + x] = acc;
    }
  }
}

// ---------------------------------------------------------------------------
// Targets
// ---------------------------------------------------------------------------
__global__ void k_zero_scalars(float* __restrict__ out, int total) {
  int i = blockIdx.x * blockDim.x + threadIdx.x;
  if (i >= total) return;
  int b   = i / (11 * Hn * Wn);
  int r   = i - b * 11 * Hn * Wn;
  int c   = r / (Hn * Wn);
  int pix = r - c * Hn * Wn;
  out[((size_t)b * 29 + 18 + c) * (Hn * Wn) + pix] = 0.0f;
}

__global__ void k_scatter(const float* __restrict__ gt, float* __restrict__ out) {
  int i = blockIdx.x * blockDim.x + threadIdx.x;
  if (i >= Bn * NBOX) return;
  int b = i >> 7, nb = i & 127;
  const float* g = gt + ((size_t)b * NBOX + nb) * 10;
  float x = g[0], y = g[1], z = g[2];
  if (fabsf(x) + fabsf(y) + fabsf(z) <= 0.0f) return;
  float gx = x * 10.0f;
  float gy = (y + 39.68f) * 10.0f;
  int gxi = (int)floorf(gx), gyi = (int)floorf(gy);
  if (gxi < 0 || gxi >= Wn || gyi < 0 || gyi >= Hn) return;
  float vals[11];
  vals[0] = gx - (float)gxi;
  vals[1] = gy - (float)gyi;
  vals[2] = z;
  vals[3] = g[3]; vals[4] = g[4]; vals[5] = g[5];
  vals[6] = __sinf(g[6]); vals[7] = __cosf(g[6]);
  vals[8] = g[8]; vals[9] = g[9];
  vals[10] = 1.0f;
  #pragma unroll
  for (int k = 0; k < 11; ++k)
    out[(((size_t)b * 29 + 18 + k) * Hn + gyi) * Wn + gxi] = vals[k];
}

__global__ void k_heatmap(const float* __restrict__ gt, float* __restrict__ out) {
  __shared__ float sx[NBOX], sy[NBOX], sinv[NBOX];
  __shared__ int   scls[NBOX];
  const int cls = blockIdx.y, b = blockIdx.z;
  const int tid = threadIdx.x;
  if (tid < NBOX) {
    const float* g = gt + ((size_t)b * NBOX + tid) * 10;
    float x = g[0], y = g[1], z = g[2], bw = g[3], bl = g[4];
    bool valid = fabsf(x) + fabsf(y) + fabsf(z) > 0.0f;
    float gx = x * 10.0f, gy = (y + 39.68f) * 10.0f;
    int gxi = (int)floorf(gx), gyi = (int)floorf(gy);
    valid = valid && gxi >= 0 && gxi < Wn && gyi >= 0 && gyi < Hn;
    int c = (int)g[7]; c = c < 0 ? 0 : (c > 3 ? 3 : c);
    float sigma = fmaxf(sqrtf(bw * bw + bl * bl) * 0.5f, 2.0f) / 3.0f;
    sx[tid] = (float)gxi;
    sy[tid] = (float)gyi;
    sinv[tid] = 1.0f / (2.0f * sigma * sigma);
    scls[tid] = valid ? c : -1;
  }
  __syncthreads();
  int pix = blockIdx.x * blockDim.x + tid;
  if (pix >= Hn * Wn) return;
  float py = (float)(pix / Wn), px = (float)(pix % Wn);
  float m = 0.0f;
  for (int k = 0; k < NBOX; ++k) {
    if (scls[k] == cls) {
      float dx = px - sx[k], dy = py - sy[k];
      m = fmaxf(m, __expf(-(dx * dx + dy * dy) * sinv[k]));
    }
  }
  out[((size_t)b * 29 + 14 + cls) * (Hn * Wn) + pix] = m;
}

// ---------------------------------------------------------------------------
// Launch
// ---------------------------------------------------------------------------
extern "C" void kernel_launch(void* const* d_in, const int* in_sizes, int n_in,
                              void* d_out, int out_size, void* d_ws, size_t ws_size,
                              hipStream_t stream) {
  const float* spatial = (const float*)d_in[0];
  const float* gt      = (const float*)d_in[1];
  const float* w1      = (const float*)d_in[2];
  const float* bng     = (const float*)d_in[3];
  const float* bnb     = (const float*)d_in[4];
  const float* bnm     = (const float*)d_in[5];
  const float* bnv     = (const float*)d_in[6];
  const float* w2      = (const float*)d_in[7];
  const float* b2      = (const float*)d_in[8];
  float* out = (float*)d_out;

  // workspace layout
  char* ws = (char*)d_ws;
  __bf16* actb  = (__bf16*)ws;                              // 4*200*200*256*2  = 81,920,000
  __bf16* wfrag = (__bf16*)(ws + 81920000);                 // 6*16*9*8*32*16*2 =  7,077,888
  float*  bnsc  = (float*)(ws + 81920000 + 7077888);        // 6*256*4
  float*  bnsh  = bnsc + 6 * Cn;

  // P1: NCHW f32 -> NHWC bf16
  {
    dim3 grid(7, 8, Bn * Hn), block(32, 8, 1);
    k_nchw_to_nhwc_bf16<<<grid, block, 0, stream>>>(spatial, actb);
  }
  // P2: weight repack into WMMA A-fragment order
  {
    int total = 6 * 16 * 9 * 8 * 32 * 16;
    k_pack_w1<<<(total + 255) / 256, 256, 0, stream>>>(w1, wfrag, total);
  }
  // P3: BN fold
  {
    int total = 6 * Cn;
    k_bn_fold<<<(total + 255) / 256, 256, 0, stream>>>(bng, bnb, bnm, bnv, bnsc, bnsh, total);
  }
  // fused conv heads -> out channels 0..13
  {
    dim3 grid((Wn + 15) / 16, Hn / 2, Bn * 6);
    k_conv_heads<<<grid, 256, 0, stream>>>(actb, wfrag, bnsc, bnsh, w2, b2, out);
  }
  // targets -> out channels 14..28
  {
    int total = Bn * 11 * Hn * Wn;
    k_zero_scalars<<<(total + 255) / 256, 256, 0, stream>>>(out, total);
    k_scatter<<<(Bn * NBOX + 255) / 256, 256, 0, stream>>>(gt, out);
    dim3 grid((Hn * Wn + 255) / 256, 4, Bn);
    k_heatmap<<<grid, 256, 0, stream>>>(gt, out);
  }
}
